// gcn_85306640433226
// MI455X (gfx1250) — matrix operand order repaired and checked by reading, not compiled, not run on hardware
//
#include <hip/hip_runtime.h>
#include <hip/hip_bf16.h>

typedef __attribute__((ext_vector_type(16))) _Float16 v16h;
typedef __attribute__((ext_vector_type(8)))  float    v8f;
typedef __attribute__((ext_vector_type(4)))  float    f4;

#define IN_DIM 128
#define H_DIM  256

// ---------------------------------------------------------------------------
// 1) Degree accumulation: one thread per edge, f32 hw atomics (L2-resident)
// ---------------------------------------------------------------------------
__global__ __launch_bounds__(256) void k_degrees(const int* __restrict__ src,
                                                 const int* __restrict__ dst,
                                                 float* __restrict__ odeg,
                                                 float* __restrict__ ideg,
                                                 int n_edges) {
    int e = blockIdx.x * 256 + threadIdx.x;
    if (e >= n_edges) return;
    unsafeAtomicAdd(&odeg[src[e]], 1.0f);
    unsafeAtomicAdd(&ideg[dst[e]], 1.0f);
}

// norm = clip(deg,1)^(-1/2)
__global__ __launch_bounds__(256) void k_norms(const float* __restrict__ odeg,
                                               const float* __restrict__ ideg,
                                               float* __restrict__ nsrc,
                                               float* __restrict__ ndst,
                                               int n_nodes) {
    int n = blockIdx.x * 256 + threadIdx.x;
    if (n >= n_nodes) return;
    nsrc[n] = rsqrtf(fmaxf(odeg[n], 1.0f));
    ndst[n] = rsqrtf(fmaxf(ideg[n], 1.0f));
}

// ---------------------------------------------------------------------------
// Pre-pack W1 (f32 [128][256], row-major KxN) into per-lane WMMA B fragments:
//   w1p[((nt*4 + ks)*32 + lane)*16 + j] = (f16) W1[(ks*32 + (lane>>4)*8 + j + (j&8))*256
//                                                  + nt*16 + (lane&15)]
// so the GEMM loads each 16-half B fragment as one contiguous 32B chunk.
// ---------------------------------------------------------------------------
__global__ __launch_bounds__(256) void k_w1_pack(const float* __restrict__ w1,
                                                 _Float16* __restrict__ w1p) {
    int t = blockIdx.x * 256 + threadIdx.x;     // 0 .. 2047
    if (t >= 16 * 4 * 32) return;
    int lane = t & 31;
    int ks   = (t >> 5) & 3;
    int nt   = t >> 7;
    int lh   = lane >> 4;
    int l15  = lane & 15;
    const float* __restrict__ colp = w1 + (size_t)(ks * 32 + lh * 8) * H_DIM
                                        + nt * 16 + l15;
    _Float16* __restrict__ op = w1p + (size_t)t * 16;
#pragma unroll
    for (int j = 0; j < 16; ++j)
        op[j] = (_Float16)colp[(size_t)(j + (j & 8)) * H_DIM];
}

// ---------------------------------------------------------------------------
// 2) Edge aggregation layer 1: one wave32 per edge; 32 lanes x float4 = 128
//    agg[dst] += x[src] * norm_src[src]   (f32 hw atomics into 25.6MB, L2-hot)
// ---------------------------------------------------------------------------
__global__ __launch_bounds__(256) void k_edge_agg1(const float* __restrict__ x,
                                                   const int* __restrict__ src,
                                                   const int* __restrict__ dst,
                                                   const float* __restrict__ nsrc,
                                                   float* __restrict__ agg,
                                                   int n_edges) {
    int wave = threadIdx.x >> 5;
    int lane = threadIdx.x & 31;
    long e = (long)blockIdx.x * 8 + wave;
    if (e >= n_edges) return;
    int s = src[e], d = dst[e];
    float ns = nsrc[s];
    f4 v = *(const f4*)(x + (size_t)s * IN_DIM + lane * 4);
    float* ap = agg + (size_t)d * IN_DIM + lane * 4;
    unsafeAtomicAdd(ap + 0, v[0] * ns);
    unsafeAtomicAdd(ap + 1, v[1] * ns);
    unsafeAtomicAdd(ap + 2, v[2] * ns);
    unsafeAtomicAdd(ap + 3, v[3] * ns);
}

// ---------------------------------------------------------------------------
// 3) WMMA GEMM:  H = relu( (agg * norm_dst[:,None]) @ W1 + b1 )   [N,256]
//    One wave per (16-row m-tile) x (64-col n-group): 4 accumulator frags,
//    A fragment reused across 4 WMMAs per k-step -> 16 WMMAs / wave.
//    A frag (16x32 f16): lane&15 = row M; lane>>4 selects K-halves
//      a[j] = A[m][lh*8 + j], a[8+j] = A[m][16 + lh*8 + j]
//    B frag: contiguous per-lane chunk from w1p (pre-packed above)
//    C/D: c[r] -> M = (lane>>4)*8 + r, N = lane&15
// ---------------------------------------------------------------------------
__global__ __launch_bounds__(256) void k_gemm1_wmma(const float* __restrict__ agg,
                                                    const float* __restrict__ ndst,
                                                    const _Float16* __restrict__ w1p,
                                                    const float* __restrict__ b1,
                                                    float* __restrict__ h,
                                                    int n_nodes) {
    const int wave = threadIdx.x >> 5;
    const int lane = threadIdx.x & 31;
    const int mtiles = (n_nodes + 15) >> 4;
    const long tile = (long)blockIdx.x * 8 + wave;   // tile = mt*4 + ng
    if (tile >= (long)mtiles * 4) return;
    const int mt = (int)(tile >> 2);
    const int ng = (int)(tile & 3);                  // n-group: columns ng*64..ng*64+63

    const int lh  = lane >> 4;                       // 0 or 1
    const int l15 = lane & 15;
    int m = mt * 16 + l15;
    if (m >= n_nodes) m = n_nodes - 1;               // clamp loads (50000%16==0 anyway)
    const float nd = ndst[m];
    const float* __restrict__ arow = agg + (size_t)m * IN_DIM;

    v8f c0 = {}, c1 = {}, c2 = {}, c3 = {};
#pragma unroll
    for (int ks = 0; ks < 4; ++ks) {
        const int kb = ks * 32 + lh * 8;
        f4 x0 = *(const f4*)(arow + kb);
        f4 x1 = *(const f4*)(arow + kb + 4);
        f4 x2 = *(const f4*)(arow + kb + 16);
        f4 x3 = *(const f4*)(arow + kb + 20);
        v16h a;
        a[0]  = (_Float16)(x0[0] * nd);  a[1]  = (_Float16)(x0[1] * nd);
        a[2]  = (_Float16)(x0[2] * nd);  a[3]  = (_Float16)(x0[3] * nd);
        a[4]  = (_Float16)(x1[0] * nd);  a[5]  = (_Float16)(x1[1] * nd);
        a[6]  = (_Float16)(x1[2] * nd);  a[7]  = (_Float16)(x1[3] * nd);
        a[8]  = (_Float16)(x2[0] * nd);  a[9]  = (_Float16)(x2[1] * nd);
        a[10] = (_Float16)(x2[2] * nd);  a[11] = (_Float16)(x2[3] * nd);
        a[12] = (_Float16)(x3[0] * nd);  a[13] = (_Float16)(x3[1] * nd);
        a[14] = (_Float16)(x3[2] * nd);  a[15] = (_Float16)(x3[3] * nd);

        // B fragments: one 32B contiguous load per (nt, ks, lane)
        const _Float16* __restrict__ bb =
            w1p + ((size_t)(ng * 4) * 4 * 32 + (size_t)ks * 32 + lane) * 16;
        v16h b0f = *(const v16h*)(bb + (size_t)0 * 4 * 32 * 16);
        v16h b1f = *(const v16h*)(bb + (size_t)1 * 4 * 32 * 16);
        v16h b2f = *(const v16h*)(bb + (size_t)2 * 4 * 32 * 16);
        v16h b3f = *(const v16h*)(bb + (size_t)3 * 4 * 32 * 16);

        c0 = __builtin_amdgcn_wmma_f32_16x16x32_f16(false, a, false, b0f, (short)0, c0, false, false);
        c1 = __builtin_amdgcn_wmma_f32_16x16x32_f16(false, a, false, b1f, (short)0, c1, false, false);
        c2 = __builtin_amdgcn_wmma_f32_16x16x32_f16(false, a, false, b2f, (short)0, c2, false, false);
        c3 = __builtin_amdgcn_wmma_f32_16x16x32_f16(false, a, false, b3f, (short)0, c3, false, false);
    }

    const int mbase = mt * 16 + lh * 8;
#pragma unroll
    for (int t = 0; t < 4; ++t) {
        v8f c = (t == 0) ? c0 : (t == 1) ? c1 : (t == 2) ? c2 : c3;
        const int ncol = (ng * 4 + t) * 16 + l15;
        const float bias = b1[ncol];
#pragma unroll
        for (int r = 0; r < 8; ++r) {
            int mr = mbase + r;
            if (mr < n_nodes) {
                float v = c[r] + bias;
                h[(size_t)mr * H_DIM + ncol] = v > 0.0f ? v : 0.0f;
            }
        }
    }
}

// ---------------------------------------------------------------------------
// 4) Layer 2 projection: y[n] = norm_src[n] * dot(h[n,:], W2)   (256 -> 1)
//    One wave per node, 32 lanes x 8 floats, shfl_xor reduce.
// ---------------------------------------------------------------------------
__global__ __launch_bounds__(256) void k_node_dot(const float* __restrict__ h,
                                                  const float* __restrict__ w2,
                                                  const float* __restrict__ nsrc,
                                                  float* __restrict__ y,
                                                  int n_nodes) {
    int wave = threadIdx.x >> 5;
    int lane = threadIdx.x & 31;
    long n = (long)blockIdx.x * 8 + wave;
    if (n >= n_nodes) return;
    const float* hp = h + (size_t)n * H_DIM + lane * 8;
    const float* wp = w2 + lane * 8;
    f4 a0 = *(const f4*)hp;       f4 a1 = *(const f4*)(hp + 4);
    f4 b0 = *(const f4*)wp;       f4 b1 = *(const f4*)(wp + 4);
    float acc = a0[0]*b0[0] + a0[1]*b0[1] + a0[2]*b0[2] + a0[3]*b0[3]
              + a1[0]*b1[0] + a1[1]*b1[1] + a1[2]*b1[2] + a1[3]*b1[3];
#pragma unroll
    for (int off = 16; off > 0; off >>= 1) acc += __shfl_xor(acc, off, 32);
    if (lane == 0) y[n] = acc * nsrc[n];
}

// ---------------------------------------------------------------------------
// 5) Edge aggregation layer 2 (scalar): agg2[dst] += y[src]
// ---------------------------------------------------------------------------
__global__ __launch_bounds__(256) void k_edge_agg2(const float* __restrict__ y,
                                                   const int* __restrict__ src,
                                                   const int* __restrict__ dst,
                                                   float* __restrict__ agg2,
                                                   int n_edges) {
    int e = blockIdx.x * 256 + threadIdx.x;
    if (e >= n_edges) return;
    unsafeAtomicAdd(&agg2[dst[e]], y[src[e]]);
}

// ---------------------------------------------------------------------------
// 6) Readout: per-graph sum of (agg2*norm_dst + b2) and node counts
// ---------------------------------------------------------------------------
__global__ __launch_bounds__(256) void k_readout(const float* __restrict__ agg2,
                                                 const float* __restrict__ ndst,
                                                 const float* __restrict__ b2,
                                                 const int* __restrict__ gid,
                                                 float* __restrict__ gsum,
                                                 float* __restrict__ gcnt,
                                                 int n_nodes) {
    int n = blockIdx.x * 256 + threadIdx.x;
    if (n >= n_nodes) return;
    float v = agg2[n] * ndst[n] + b2[0];
    int g = gid[n];
    unsafeAtomicAdd(&gsum[g], v);
    unsafeAtomicAdd(&gcnt[g], 1.0f);
}

__global__ __launch_bounds__(256) void k_finalize(const float* __restrict__ gsum,
                                                  const float* __restrict__ gcnt,
                                                  float* __restrict__ out,
                                                  int n_graphs) {
    int g = blockIdx.x * 256 + threadIdx.x;
    if (g >= n_graphs) return;
    out[g] = gsum[g] / fmaxf(gcnt[g], 1.0f);
}

// ---------------------------------------------------------------------------
extern "C" void kernel_launch(void* const* d_in, const int* in_sizes, int n_in,
                              void* d_out, int out_size, void* d_ws, size_t ws_size,
                              hipStream_t stream) {
    const float* x   = (const float*)d_in[0];
    const int*   src = (const int*)  d_in[1];
    const int*   dst = (const int*)  d_in[2];
    const int*   gid = (const int*)  d_in[3];
    const float* W1  = (const float*)d_in[4];
    const float* b1  = (const float*)d_in[5];
    const float* W2  = (const float*)d_in[6];
    const float* b2  = (const float*)d_in[7];

    const int n_nodes  = in_sizes[0] / IN_DIM;   // 50000
    const int n_edges  = in_sizes[1];            // 1600000
    const int n_graphs = out_size;               // 64
    float* out = (float*)d_out;

    // ---- workspace layout (floats) ----
    float* ws = (float*)d_ws;
    size_t o = 0;
    float* odeg = ws + o;  o += n_nodes;
    float* ideg = ws + o;  o += n_nodes;
    float* nsrc = ws + o;  o += n_nodes;
    float* ndst = ws + o;  o += n_nodes;
    float* y    = ws + o;  o += n_nodes;
    float* agg2 = ws + o;  o += n_nodes;
    float* gsum = ws + o;  o += n_graphs;
    float* gcnt = ws + o;  o += n_graphs;
    o = (o + 7) & ~(size_t)7;                       // 32B align for v16h loads
    _Float16* w1p = (_Float16*)(ws + o); o += (IN_DIM * H_DIM) / 2;
    float* agg1 = ws + o;  o += (size_t)n_nodes * IN_DIM;
    float* h    = ws + o;  o += (size_t)n_nodes * H_DIM;
    // total ~78 MB

    // ---- zero accumulators (graph-capture-safe memset nodes) ----
    hipMemsetAsync(odeg, 0, sizeof(float) * 2 * n_nodes, stream);              // odeg+ideg
    hipMemsetAsync(agg2, 0, sizeof(float) * ((size_t)n_nodes + 2 * n_graphs), stream); // agg2+gsum+gcnt
    hipMemsetAsync(agg1, 0, sizeof(float) * (size_t)n_nodes * IN_DIM, stream);

    // ---- pipeline ----
    k_degrees<<<(n_edges + 255) / 256, 256, 0, stream>>>(src, dst, odeg, ideg, n_edges);
    k_norms  <<<(n_nodes + 255) / 256, 256, 0, stream>>>(odeg, ideg, nsrc, ndst, n_nodes);
    k_w1_pack<<<(16 * 4 * 32 + 255) / 256, 256, 0, stream>>>(W1, w1p);

    k_edge_agg1<<<(n_edges + 7) / 8, 256, 0, stream>>>(x, src, dst, nsrc, agg1, n_edges);

    {
        long mtiles = (n_nodes + 15) / 16;
        long wtiles = mtiles * 4;           // 4 n-groups of 64 columns each
        k_gemm1_wmma<<<(unsigned)((wtiles + 7) / 8), 256, 0, stream>>>(
            agg1, ndst, w1p, b1, h, n_nodes);
    }

    k_node_dot <<<(n_nodes + 7) / 8, 256, 0, stream>>>(h, W2, nsrc, y, n_nodes);
    k_edge_agg2<<<(n_edges + 255) / 256, 256, 0, stream>>>(y, src, dst, agg2, n_edges);
    k_readout  <<<(n_nodes + 255) / 256, 256, 0, stream>>>(agg2, ndst, b2, gid, gsum, gcnt, n_nodes);
    k_finalize <<<(n_graphs + 255) / 256, 256, 0, stream>>>(gsum, gcnt, out, n_graphs);
}
